// CrossAttention1D_14577119002773
// MI455X (gfx1250) — matrix operand order repaired and checked by reading, not compiled, not run on hardware
//
#include <hip/hip_runtime.h>
#include <hip/hip_bf16.h>

// ---------------- types ----------------
typedef __bf16 bhalf;
typedef __bf16 v16bf __attribute__((ext_vector_type(16)));
typedef float  v8f   __attribute__((ext_vector_type(8)));
typedef unsigned int u32x4 __attribute__((ext_vector_type(4)));
typedef int v4i __attribute__((ext_vector_type(4)));

// problem constants (fixed by the reference harness)
#define BSZ  4
#define CQ_  1024
#define LQ_  2048
#define H_   16
#define DH_  64

// CDNA5 async global->LDS DMA (ASYNCcnt path). Guarded so the file compiles
// even if a toolchain doesn't expose the builtins. (Probe-confirmed present:
// signature is (global v4i*, lds v4i*, int offset, int cpol).)
#if defined(__has_builtin)
#  if __has_builtin(__builtin_amdgcn_global_load_async_to_lds_b128) && \
      __has_builtin(__builtin_amdgcn_s_wait_asynccnt)
#    define HAVE_ASYNC_LDS 1
#  endif
#endif
#ifndef HAVE_ASYNC_LDS
#  define HAVE_ASYNC_LDS 0
#endif

#if HAVE_ASYNC_LDS
typedef __attribute__((address_space(1))) v4i* as1_v4i;
typedef __attribute__((address_space(3))) v4i* as3_v4i;
#endif

__device__ __forceinline__ void wait_async_lds() {
#if HAVE_ASYNC_LDS
  __builtin_amdgcn_s_wait_asynccnt(0);
#endif
}

__device__ __forceinline__ v8f zero8() {
  v8f z;
#pragma unroll
  for (int i = 0; i < 8; ++i) z[i] = 0.0f;
  return z;
}

// A-matrix (16x32 bf16) fragment from an LDS row (ISA 7.12.2):
// lanes 0-15: V0..3 hold K=0..7,  V4..7 hold K=16..23
// lanes16-31: V0..3 hold K=8..15, V4..7 hold K=24..31
__device__ __forceinline__ v16bf lds_frag_a(const bhalf* row, int hf, int k0) {
  union { v16bf v; u32x4 q[2]; } f;
  const bhalf* p = row + k0 + (hf ? 8 : 0);
  f.q[0] = *(const u32x4*)(p);
  f.q[1] = *(const u32x4*)(p + 16);
  return f.v;
}

// B-matrix (32x16 bf16) fragment: lanes 0-15 K=0..7,16..23; lanes 16-31 K=8..15,24..31
__device__ __forceinline__ v16bf lds_frag_b(const bhalf* row, int hf, int k0) {
  union { v16bf v; u32x4 q[2]; } f;
  const bhalf* p = row + k0 + (hf ? 8 : 0);
  f.q[0] = *(const u32x4*)(p);
  f.q[1] = *(const u32x4*)(p + 16);
  return f.v;
}

// ---------------- fp32 -> bf16 convert ----------------
__global__ void cvt_bf16_k(const float* __restrict__ src, bhalf* __restrict__ dst, int n) {
  int i = blockIdx.x * blockDim.x + threadIdx.x;
  if (i < n) dst[i] = (bhalf)src[i];
}

// ======================= GEMM helpers (tile = 128x128x64) =======================
// X tile (64k x 128n) -> regs (16 dwords/thread), later stored transposed [n][k]
__device__ __forceinline__ void load_x_regs(const bhalf* __restrict__ Xb, int kt,
                                            int n0, int N, int tid, unsigned xr[16]) {
#pragma unroll
  for (int t = 0; t < 16; ++t) {
    int idx = tid + t * 256;
    int k = idx >> 6, n2 = (idx & 63) << 1;
    xr[t] = *(const unsigned*)&Xb[(size_t)(kt + k) * N + n0 + n2];
  }
}
__device__ __forceinline__ void store_x_lds(bhalf Xt[][72], int tid, const unsigned xr[16]) {
#pragma unroll
  for (int t = 0; t < 16; ++t) {
    int idx = tid + t * 256;
    int k = idx >> 6, n2 = (idx & 63) << 1;
    Xt[n2][k]     = ((const bhalf*)&xr[t])[0];
    Xt[n2 + 1][k] = ((const bhalf*)&xr[t])[1];
  }
}

#if HAVE_ASYNC_LDS
// W tile (128m x 64k): no transpose needed -> per-lane 16B async DMA into LDS
__device__ __forceinline__ void async_w_tile(const bhalf* __restrict__ W, int m0, int K,
                                             int kt, bhalf Wl[][72], int tid) {
#pragma unroll
  for (int t = 0; t < 4; ++t) {
    int idx = tid + t * 256;
    int r = idx >> 3, c8 = (idx & 7) << 3;
    __builtin_amdgcn_global_load_async_to_lds_b128(
        (as1_v4i)(bhalf*)(W + (size_t)(m0 + r) * K + kt + c8),
        (as3_v4i)&Wl[r][c8], 0, 0);
  }
}
#else
__device__ __forceinline__ void load_w_regs(const bhalf* __restrict__ W, int m0, int K,
                                            int kt, int tid, unsigned wr[16]) {
#pragma unroll
  for (int t = 0; t < 16; ++t) {
    int idx = tid + t * 256;
    int r = idx >> 5, c2 = (idx & 31) << 1;
    wr[t] = *(const unsigned*)&W[(size_t)(m0 + r) * K + kt + c2];
  }
}
__device__ __forceinline__ void store_w_lds(bhalf Wl[][72], int tid, const unsigned wr[16]) {
#pragma unroll
  for (int t = 0; t < 16; ++t) {
    int idx = tid + t * 256;
    int r = idx >> 5, c2 = (idx & 31) << 1;
    *(unsigned*)&Wl[r][c2] = wr[t];
  }
}
#endif

// ---------------- bf16 GEMM: C[b] = W(MxK) * X[b](KxN) ----------------
// outB != nullptr : store bf16 (projection path)
// outB == nullptr : store fp32 + bias + residual (output projection path)
__global__ __launch_bounds__(256) void gemm_wmma_bf16(
    const bhalf* __restrict__ W, const bhalf* __restrict__ X,
    bhalf* __restrict__ outB, float* __restrict__ outF,
    const float* __restrict__ bias, const float* __restrict__ resid,
    int M, int K, int N)
{
  __shared__ __align__(16) bhalf Wl[2][128][72];  // ping-pong W tiles  [m][k]
  __shared__ __align__(16) bhalf Xt[2][128][72];  // ping-pong X tiles  [n][k]

  const int b   = blockIdx.z;
  const int m0  = blockIdx.y * 128;
  const int n0  = blockIdx.x * 128;
  const int tid = threadIdx.x;
  const int lane = tid & 31;
  const int w    = tid >> 5;          // 8 waves
  const int hf   = lane >> 4;
  const int ln   = lane & 15;
  const int wm   = (w & 1) * 64;      // 2 waves along M
  const int wn   = (w >> 1) * 32;     // 4 waves along N

  const bhalf* Xb = X + (size_t)b * K * N;

  v8f acc[4][2];
#pragma unroll
  for (int i = 0; i < 4; ++i)
#pragma unroll
    for (int j = 0; j < 2; ++j) acc[i][j] = zero8();

  unsigned xreg[16];
#if !HAVE_ASYNC_LDS
  unsigned wreg[16];
#endif

  // ---- prologue: stage tile 0 into buffer 0 ----
  load_x_regs(Xb, 0, n0, N, tid, xreg);
#if HAVE_ASYNC_LDS
  async_w_tile(W, m0, K, 0, Wl[0], tid);
#else
  load_w_regs(W, m0, K, 0, tid, wreg);
  store_w_lds(Wl[0], tid, wreg);
#endif
  store_x_lds(Xt[0], tid, xreg);

  const int NIT = K / 64;
  for (int it = 0; it < NIT; ++it) {
    const int cur = it & 1;
    wait_async_lds();          // my async W writes into buf[cur] are done
    __syncthreads();           // publish buf[cur]; everyone done reading buf[cur^1]

    if (it + 1 < NIT) {        // start tile it+1 while we compute tile it
      load_x_regs(Xb, (it + 1) * 64, n0, N, tid, xreg);
#if HAVE_ASYNC_LDS
      async_w_tile(W, m0, K, (it + 1) * 64, Wl[cur ^ 1], tid);
#else
      load_w_regs(W, m0, K, (it + 1) * 64, tid, wreg);
#endif
    }

    // ---- 16 WMMAs on buffer `cur` ----
#pragma unroll
    for (int kd = 0; kd < 64; kd += 32) {
      v16bf af[4], bfr[2];
#pragma unroll
      for (int mb = 0; mb < 4; ++mb)
        af[mb] = lds_frag_a(&Wl[cur][wm + mb * 16 + ln][0], hf, kd);
#pragma unroll
      for (int nb = 0; nb < 2; ++nb)
        bfr[nb] = lds_frag_b(&Xt[cur][wn + nb * 16 + ln][0], hf, kd);
#pragma unroll
      for (int mb = 0; mb < 4; ++mb)
#pragma unroll
        for (int nb = 0; nb < 2; ++nb)
          acc[mb][nb] = __builtin_amdgcn_wmma_f32_16x16x32_bf16(
              false, af[mb], false, bfr[nb], (short)0, acc[mb][nb], false, false);
    }

    if (it + 1 < NIT) {        // stage next tile into the other buffer
      store_x_lds(Xt[cur ^ 1], tid, xreg);
#if !HAVE_ASYNC_LDS
      store_w_lds(Wl[cur ^ 1], tid, wreg);
#endif
    }
  }

  // epilogue: C/D layout -> row = r + 8*hf, col = lane&15
#pragma unroll
  for (int mb = 0; mb < 4; ++mb)
#pragma unroll
    for (int nb = 0; nb < 2; ++nb) {
      int col = n0 + wn + nb * 16 + ln;
#pragma unroll
      for (int r = 0; r < 8; ++r) {
        int row = m0 + wm + mb * 16 + r + (hf ? 8 : 0);
        size_t o = ((size_t)b * M + row) * N + col;
        float v = acc[mb][nb][r];
        if (outB) outB[o] = (bhalf)v;
        else      outF[o] = v + bias[row] + resid[o];
      }
    }
}

// ======================= attention helpers =======================
__device__ __forceinline__ void load_k_regs(const bhalf* __restrict__ Kh, int j0,
                                            int tid, unsigned kr[8]) {
#pragma unroll
  for (int t = 0; t < 8; ++t) {
    int idx = tid + t * 256;
    int d = idx >> 5, j2 = (idx & 31) << 1;
    kr[t] = *(const unsigned*)&Kh[(size_t)d * LQ_ + j0 + j2];
  }
}
__device__ __forceinline__ void store_k_lds(bhalf Kl[][72], int tid, const unsigned kr[8]) {
#pragma unroll
  for (int t = 0; t < 8; ++t) {
    int idx = tid + t * 256;
    int d = idx >> 5, j2 = (idx & 31) << 1;
    Kl[j2][d]     = ((const bhalf*)&kr[t])[0];   // transposed: [j][d]
    Kl[j2 + 1][d] = ((const bhalf*)&kr[t])[1];
  }
}

#if HAVE_ASYNC_LDS
__device__ __forceinline__ void async_v_tile(const bhalf* __restrict__ Vh, int j0,
                                             bhalf Vl[][72], int tid) {
#pragma unroll
  for (int t = 0; t < 2; ++t) {
    int idx = tid + t * 256;
    int d = idx >> 3, j8 = (idx & 7) << 3;
    __builtin_amdgcn_global_load_async_to_lds_b128(
        (as1_v4i)(bhalf*)(Vh + (size_t)d * LQ_ + j0 + j8),
        (as3_v4i)&Vl[d][j8], 0, 0);
  }
}
#else
__device__ __forceinline__ void load_v_regs(const bhalf* __restrict__ Vh, int j0,
                                            int tid, unsigned vr[8]) {
#pragma unroll
  for (int t = 0; t < 8; ++t) {
    int idx = tid + t * 256;
    int d = idx >> 5, j2 = (idx & 31) << 1;
    vr[t] = *(const unsigned*)&Vh[(size_t)d * LQ_ + j0 + j2];
  }
}
__device__ __forceinline__ void store_v_lds(bhalf Vl[][72], int tid, const unsigned vr[8]) {
#pragma unroll
  for (int t = 0; t < 8; ++t) {
    int idx = tid + t * 256;
    int d = idx >> 5, j2 = (idx & 31) << 1;
    *(unsigned*)&Vl[d][j2] = vr[t];              // natural: [d][j]
  }
}
#endif

// ---------------- flash attention (per b,h; 128 query rows / block) ----------------
__global__ __launch_bounds__(256) void attn_wmma(
    const bhalf* __restrict__ Q, const bhalf* __restrict__ Kmat,
    const bhalf* __restrict__ V, bhalf* __restrict__ O)
{
  __shared__ __align__(16) bhalf Ql[128][72];     // Ql[i][d]  (A for S), pre-scaled
  __shared__ __align__(16) bhalf Kl[2][64][72];   // Kl[j][d]  (B for S) ping-pong
  __shared__ __align__(16) bhalf Vl[2][64][72];   // Vl[d][j]  (B for O) ping-pong
  __shared__ __align__(16) bhalf Pl[128][72];     // Pl[i][j]  (A for O)

  const int i0 = blockIdx.x * 128;
  const int h  = blockIdx.y;
  const int b  = blockIdx.z;
  const int tid  = threadIdx.x;
  const int lane = tid & 31;
  const int w    = tid >> 5;         // 8 waves, each owns 16 query rows
  const int hf   = lane >> 4;
  const int ln   = lane & 15;
  const int iw   = w * 16;

  const size_t headBase = ((size_t)(b * H_ + h)) * DH_ * (size_t)LQ_;
  const bhalf* Qh = Q + headBase;
  const bhalf* Kh = Kmat + headBase;
  const bhalf* Vh = V + headBase;

  // softmax scale folded into Q, in log2 domain: D^-0.5 * log2(e)
  const float QSCL = 0.125f * 1.44269504088896f;

  // load Q tile (64 d x 128 i), store transposed Ql[i][d], pre-scaled
#pragma unroll
  for (int t = 0; t < 16; ++t) {
    int idx = tid + t * 256;              // 4096 dwords
    int d = idx >> 6, i2 = (idx & 63) << 1;
    unsigned u = *(const unsigned*)&Qh[(size_t)d * LQ_ + i0 + i2];
    Ql[i2][d]     = (bhalf)((float)((const bhalf*)&u)[0] * QSCL);
    Ql[i2 + 1][d] = (bhalf)((float)((const bhalf*)&u)[1] * QSCL);
  }

  v8f oacc[4];
#pragma unroll
  for (int db = 0; db < 4; ++db) oacc[db] = zero8();
  float mrun[8], lrun[8];
#pragma unroll
  for (int r = 0; r < 8; ++r) { mrun[r] = -1e30f; lrun[r] = 0.0f; }

  unsigned kreg[8];
#if !HAVE_ASYNC_LDS
  unsigned vreg[8];
#endif

  // ---- prologue: stage j-tile 0 into buffer 0 ----
  load_k_regs(Kh, 0, tid, kreg);
#if HAVE_ASYNC_LDS
  async_v_tile(Vh, 0, Vl[0], tid);
#else
  load_v_regs(Vh, 0, tid, vreg);
  store_v_lds(Vl[0], tid, vreg);
#endif
  store_k_lds(Kl[0], tid, kreg);

  for (int it = 0; it < LQ_ / 64; ++it) {
    const int bsel = it & 1;
    const int j0n  = (it + 1) * 64;
    wait_async_lds();          // my async V writes into buf[bsel] are done
    __syncthreads();           // publish K/V buf[bsel]; all done with buf[bsel^1]

    if (j0n < LQ_) {           // start next j-tile during S + softmax + O
      load_k_regs(Kh, j0n, tid, kreg);
#if HAVE_ASYNC_LDS
      async_v_tile(Vh, j0n, Vl[bsel ^ 1], tid);
#else
      load_v_regs(Vh, j0n, tid, vreg);
#endif
    }

    // S = (Q^T) K for this wave's 16 rows x 64 cols (log2-domain scores)
    v8f s[4];
#pragma unroll
    for (int jb = 0; jb < 4; ++jb) {
      s[jb] = zero8();
#pragma unroll
      for (int kd = 0; kd < 64; kd += 32) {
        v16bf aq = lds_frag_a(&Ql[iw + ln][0], hf, kd);
        v16bf bk = lds_frag_b(&Kl[bsel][jb * 16 + ln][0], hf, kd);
        s[jb] = __builtin_amdgcn_wmma_f32_16x16x32_bf16(
            false, aq, false, bk, (short)0, s[jb], false, false);
      }
    }

    // online softmax (row r+8*hf lives across one 16-lane half)
    float tmax[8];
#pragma unroll
    for (int r = 0; r < 8; ++r) {
      float m = s[0][r];
#pragma unroll
      for (int jb = 1; jb < 4; ++jb) m = fmaxf(m, s[jb][r]);
#pragma unroll
      for (int msk = 1; msk < 16; msk <<= 1)
        m = fmaxf(m, __shfl_xor(m, msk, 32));
      tmax[r] = m;
    }
    float alpha[8], rsum[8];
#pragma unroll
    for (int r = 0; r < 8; ++r) {
      float nm = fmaxf(mrun[r], tmax[r]);
      alpha[r] = exp2f(mrun[r] - nm);
      mrun[r]  = nm;
      rsum[r]  = 0.0f;
    }
#pragma unroll
    for (int jb = 0; jb < 4; ++jb)
#pragma unroll
      for (int r = 0; r < 8; ++r) {
        float p = exp2f(s[jb][r] - mrun[r]);
        s[jb][r] = p;
        rsum[r] += p;
      }
#pragma unroll
    for (int r = 0; r < 8; ++r) {
      float t = rsum[r];
#pragma unroll
      for (int msk = 1; msk < 16; msk <<= 1) t += __shfl_xor(t, msk, 32);
      lrun[r] = lrun[r] * alpha[r] + t;
    }
#pragma unroll
    for (int db = 0; db < 4; ++db)
#pragma unroll
      for (int r = 0; r < 8; ++r) oacc[db][r] *= alpha[r];

    // write P (C layout) to LDS as bf16 A-operand source
#pragma unroll
    for (int jb = 0; jb < 4; ++jb)
#pragma unroll
      for (int r = 0; r < 8; ++r)
        Pl[iw + r + (hf ? 8 : 0)][jb * 16 + ln] = (bhalf)s[jb][r];
    __syncthreads();

    // O += P * V^T   (A = P 16x64, B[k=j][n=d] = Vl[d][j])
#pragma unroll
    for (int db = 0; db < 4; ++db)
#pragma unroll
      for (int kj = 0; kj < 64; kj += 32) {
        v16bf ap = lds_frag_a(&Pl[iw + ln][0], hf, kj);
        v16bf bv = lds_frag_b(&Vl[bsel][db * 16 + ln][0], hf, kj);
        oacc[db] = __builtin_amdgcn_wmma_f32_16x16x32_bf16(
            false, ap, false, bv, (short)0, oacc[db], false, false);
      }

    if (j0n < LQ_) {           // stage next tile into the other buffer
      store_k_lds(Kl[bsel ^ 1], tid, kreg);
#if !HAVE_ASYNC_LDS
      store_v_lds(Vl[bsel ^ 1], tid, vreg);
#endif
    }
  }

  // normalize and store O[b, h*64+d, i] (bf16 for the Wo GEMM)
  float inv[8];
#pragma unroll
  for (int r = 0; r < 8; ++r) inv[r] = 1.0f / lrun[r];
#pragma unroll
  for (int db = 0; db < 4; ++db) {
    int d = db * 16 + ln;
#pragma unroll
    for (int r = 0; r < 8; ++r) {
      int i = i0 + iw + r + (hf ? 8 : 0);
      O[headBase + (size_t)d * LQ_ + i] = (bhalf)(oacc[db][r] * inv[r]);
    }
  }
}

// ---------------- host-side launch ----------------
extern "C" void kernel_launch(void* const* d_in, const int* in_sizes, int n_in,
                              void* d_out, int out_size, void* d_ws, size_t ws_size,
                              hipStream_t stream) {
  const float* query   = (const float*)d_in[0];
  const float* context = (const float*)d_in[1];
  const float* Wq = (const float*)d_in[2];
  const float* Wk = (const float*)d_in[3];
  const float* Wv = (const float*)d_in[4];
  const float* Wo = (const float*)d_in[5];
  const float* bo = (const float*)d_in[6];
  float* out = (float*)d_out;

  const size_t actN = (size_t)BSZ * CQ_ * LQ_;   // 8.4M elems
  const size_t wN   = (size_t)CQ_ * CQ_;         // 1M elems

  char* p = (char*)d_ws;
  bhalf* xq = (bhalf*)p; p += actN * sizeof(bhalf);
  bhalf* xc = (bhalf*)p; p += actN * sizeof(bhalf);
  bhalf* wq = (bhalf*)p; p += wN * sizeof(bhalf);
  bhalf* wk = (bhalf*)p; p += wN * sizeof(bhalf);
  bhalf* wv = (bhalf*)p; p += wN * sizeof(bhalf);
  bhalf* wo = (bhalf*)p; p += wN * sizeof(bhalf);
  bhalf* Qb = (bhalf*)p; p += actN * sizeof(bhalf);
  bhalf* Kb = (bhalf*)p; p += actN * sizeof(bhalf);
  bhalf* Vb = (bhalf*)p; p += actN * sizeof(bhalf);
  bhalf* Ob = (bhalf*)p; p += actN * sizeof(bhalf);

  // 1) convert everything to bf16
  cvt_bf16_k<<<(actN + 255) / 256, 256, 0, stream>>>(query,   xq, (int)actN);
  cvt_bf16_k<<<(actN + 255) / 256, 256, 0, stream>>>(context, xc, (int)actN);
  cvt_bf16_k<<<(wN + 255) / 256,   256, 0, stream>>>(Wq, wq, (int)wN);
  cvt_bf16_k<<<(wN + 255) / 256,   256, 0, stream>>>(Wk, wk, (int)wN);
  cvt_bf16_k<<<(wN + 255) / 256,   256, 0, stream>>>(Wv, wv, (int)wN);
  cvt_bf16_k<<<(wN + 255) / 256,   256, 0, stream>>>(Wo, wo, (int)wN);

  // 2) projections: Q = Wq*xq, K = Wk*xc, V = Wv*xc
  dim3 ggrid(LQ_ / 128, CQ_ / 128, BSZ);
  gemm_wmma_bf16<<<ggrid, 256, 0, stream>>>(wq, xq, Qb, nullptr, nullptr, nullptr, CQ_, CQ_, LQ_);
  gemm_wmma_bf16<<<ggrid, 256, 0, stream>>>(wk, xc, Kb, nullptr, nullptr, nullptr, CQ_, CQ_, LQ_);
  gemm_wmma_bf16<<<ggrid, 256, 0, stream>>>(wv, xc, Vb, nullptr, nullptr, nullptr, CQ_, CQ_, LQ_);

  // 3) flash attention -> Ob
  dim3 agrid(LQ_ / 128, H_, BSZ);
  attn_wmma<<<agrid, 256, 0, stream>>>(Qb, Kb, Vb, Ob);

  // 4) output projection + bias + residual -> fp32 out
  gemm_wmma_bf16<<<ggrid, 256, 0, stream>>>(wo, Ob, nullptr, out, bo, query, CQ_, CQ_, LQ_);
}
